// TimeLSTM_24824910971304
// MI455X (gfx1250) — matrix operand order repaired and checked by reading, not compiled, not run on hardware
//
#include <hip/hip_runtime.h>
#include <hip/hip_bf16.h>
#include <cstdint>
#include <cstddef>

// Problem dims (fixed by the reference)
#define Bb   256
#define Ss   512
#define INN  512
#define HH   512
#define G4   2048   // 4*H

typedef __bf16 v16bf __attribute__((ext_vector_type(16)));
typedef float  v8f   __attribute__((ext_vector_type(8)));

union BFPack {
    v16bf v;
    uint32_t u[8];
    unsigned short s[16];
    uint4 q[2];
};

__device__ __forceinline__ unsigned short f2bf(float f) {
    uint32_t u = __float_as_uint(f);
    u += 0x7FFFu + ((u >> 16) & 1u);       // round-to-nearest-even
    return (unsigned short)(u >> 16);
}
// native-HW transcendentals: v_exp_f32 / v_rcp_f32
__device__ __forceinline__ float fsigm(float x) {
    return __builtin_amdgcn_rcpf(1.0f + __builtin_amdgcn_exp2f(x * -1.4426950408889634f));
}
__device__ __forceinline__ float ftanh(float x) {
    float e = __builtin_amdgcn_exp2f(x * 2.8853900817779268f);   // e^(2x)
    return 1.0f - 2.0f * __builtin_amdgcn_rcpf(e + 1.0f);
}

// ---------------------------------------------------------------------------
// Kernel 1: one-time weight conversion fp32 -> bf16 (row-major [g][k], so a
// WMMA B-operand lane reads 32 contiguous bytes), plus combined gate bias.
// ---------------------------------------------------------------------------
__global__ void __launch_bounds__(256) prep_kernel(
    const float* __restrict__ Wall, const float* __restrict__ Uw,
    const float* __restrict__ Wd,   const float* __restrict__ Wall_b,
    const float* __restrict__ U_b,
    unsigned short* __restrict__ wallbf, unsigned short* __restrict__ ubf,
    unsigned short* __restrict__ wdbf,   float* __restrict__ bias)
{
    int idx = blockIdx.x * blockDim.x + threadIdx.x;
    if (idx < G4 * HH) { wallbf[idx] = f2bf(Wall[idx]); ubf[idx] = f2bf(Uw[idx]); }
    if (idx < HH * HH) { wdbf[idx] = f2bf(Wd[idx]); }
    if (idx < G4)      { bias[idx] = Wall_b[idx] + U_b[idx]; }
}

// ---------------------------------------------------------------------------
// Kernel 2: persistent scan. 16 workgroups (16-row batch tiles), 512 threads =
// 16 wave32; wave w owns hidden columns [32w, 32w+32).
// ---------------------------------------------------------------------------
#define XPITCH   516                       // floats per LDS x row (512 + 4 pad)
#define XBUFB    (16 * XPITCH * 4)         // 33024 bytes per f32 x buffer
#define HPITCH   520                       // ushorts per LDS bf16 row (512 + 8 pad)
#define HROWB    (16 * HPITCH * 2)         // 16640 bytes
#define LDS_XBF  (2 * XBUFB)               // 66048
#define LDS_H    (LDS_XBF + HROWB)         // 82688
#define LDS_C    (LDS_H + HROWB)           // 99328
#define SMEM_BYTES (LDS_C + HROWB)         // 115968

#define WMMA_BF16(a, b, c) \
    __builtin_amdgcn_wmma_f32_16x16x32_bf16(false, (a), false, (b), (short)0, (c), false, false)

__global__ void __launch_bounds__(512, 1) scan_kernel(
    const float* __restrict__ inputs,         // [B,S,IN] fp32
    const float* __restrict__ tstamps,        // [B,S]
    const unsigned short* __restrict__ Wall,  // [2048][512] bf16
    const unsigned short* __restrict__ Uw,    // [2048][512] bf16
    const unsigned short* __restrict__ Wd,    // [512][512]  bf16
    const float* __restrict__ bias,           // W_all_b + U_all_b [2048]
    const float* __restrict__ Wd_b,           // [512]
    float* __restrict__ out,                  // [B,S,H] (o gate)
    float* __restrict__ hout,                 // [B,H]
    float* __restrict__ cout_)                // [B,H]
{
    extern __shared__ unsigned char smem[];
    float*          xf  = (float*)smem;                       // f32 staging, 2 bufs
    unsigned short* xbf = (unsigned short*)(smem + LDS_XBF);  // bf16 x operand copy
    unsigned short* hbf = (unsigned short*)(smem + LDS_H);
    unsigned short* cbf = (unsigned short*)(smem + LDS_C);

    const int tid  = threadIdx.x;
    const int wave = tid >> 5;
    const int lane = tid & 31;
    const int lh   = lane & 15;
    const int hi   = lane >> 4;
    const int bm0  = blockIdx.x * 16;
    const unsigned smem_base = (unsigned)(uintptr_t)(void*)smem; // LDS byte offset

    // zero bf16 recurrent state
    for (int i = tid; i < 16 * HPITCH; i += 512) { hbf[i] = 0; cbf[i] = 0; }

    // ---- async-copy bases: 4 segments of 16B per thread per step -----------
    uint64_t gxb[4];
    unsigned lxb[4];
#pragma unroll
    for (int qq = 0; qq < 4; ++qq) {
        int s   = tid + 512 * qq;       // 0..2047 16B segments
        int row = s >> 7;               // 0..15
        int seg = s & 127;
        lxb[qq] = (unsigned)(row * (XPITCH * 4) + seg * 16);
        gxb[qq] = (uint64_t)(uintptr_t)(inputs + (size_t)(bm0 + row) * Ss * INN + seg * 4);
    }
    auto issue_x = [&](int t, int buf) {
        const unsigned tb = (unsigned)t * (INN * 4u);
        const unsigned lb = smem_base + (unsigned)(buf * XBUFB);
#pragma unroll
        for (int qq = 0; qq < 4; ++qq) {
            unsigned lds = lb + lxb[qq];
            uint64_t ga  = gxb[qq] + tb;
            asm volatile("global_load_async_to_lds_b128 %0, %1, off"
                         :: "v"(lds), "v"(ga) : "memory");
        }
    };
    issue_x(0, 0);

    // ---- per-wave constants -------------------------------------------------
    float bF[2], bI[2], bO[2], bC[2], bD[2];
#pragma unroll
    for (int j = 0; j < 2; ++j) {
        int n = wave * 32 + 16 * j + lh;
        bF[j] = bias[n];        bI[j] = bias[512 + n];
        bO[j] = bias[1024 + n]; bC[j] = bias[1536 + n];
        bD[j] = Wd_b[n];
    }
    // 32-bit weight row BYTE offsets for the 10 distinct tile rows.
    // Tiles 0..7: gate qg=i&3, sub j=i>>2.  Tiles 8..9: W_d rows.  Tiles 10..17
    // reuse roffB[i-10] against the U matrix.
    unsigned roffB[10];
#pragma unroll
    for (int i = 0; i < 8; ++i) {
        unsigned qg = i & 3, j = i >> 2;
        roffB[i] = (qg * 512u + (unsigned)(wave * 32 + 16 * j + lh)) << 10;
    }
#pragma unroll
    for (int i = 8; i < 10; ++i)
        roffB[i] = ((unsigned)(wave * 32 + 16 * (i - 8) + lh)) << 10;

    const v8f vzero = {0.f, 0.f, 0.f, 0.f, 0.f, 0.f, 0.f, 0.f};
    v8f creg[2];                 // f32 master copy of c (accumulator layout)
    creg[0] = vzero; creg[1] = vzero;

    asm volatile("s_wait_asynccnt 0" ::: "memory");
    __syncthreads();

    const unsigned short* hrow = hbf + lh * HPITCH;
    const unsigned short* crow = cbf + lh * HPITCH;
    const unsigned short* xrow = xbf + lh * HPITCH;
    const char* WallB = (const char*)Wall;
    const char* UwB   = (const char*)Uw;
    const char* WdB   = (const char*)Wd;

    for (int t = 0; t < Ss; ++t) {
        const int buf = t & 1;
        if (t + 1 < Ss) issue_x(t + 1, buf ^ 1);   // overlap next-step fetch

        // ---- cooperative f32 -> bf16 conversion of this step's x tile ------
        {
            const int row = tid >> 5;
            const int c0  = (tid & 31) * 16;
            const float* src = xf + buf * (16 * XPITCH) + row * XPITCH + c0;
            unsigned short* dst = xbf + row * HPITCH + c0;
            BFPack tv;
#pragma unroll
            for (int e = 0; e < 16; ++e) tv.s[e] = f2bf(src[e]);
            *(uint4*)(dst)     = tv.q[0];
            *(uint4*)(dst + 8) = tv.q[1];
        }
        __syncthreads();

        // ---- K-loop: 18 WMMA per k-step, continuous cross-k-step pipeline --
        v8f acc[10];               // 0..7 gates (qg + 4*j), 8..9 c_s1
#pragma unroll
        for (int g = 0; g < 10; ++g) acc[g] = vzero;

        auto loadt = [&](int i, unsigned kbb, BFPack& dst) {
            const char* mp = (i < 8) ? WallB : ((i < 10) ? WdB : UwB);
            unsigned boff = roffB[(i < 10) ? i : (i - 10)] + kbb;
            // keep the offset as a fresh 32-bit VGPR so the backend selects the
            // SADDR + zext(voffset32) global-load form (no 64-bit addr pairs)
            asm("" : "+v"(boff));
            const uint4* p = (const uint4*)(mp + boff);
            dst.q[0] = p[0]; dst.q[1] = p[1];
        };

        // 18 % 6 == 0, so the rotating-buffer index is continuous across
        // k-steps: tail loads of step k feed tiles 0..2 of step k+1.
        BFPack bb[6];
        {
            const unsigned kbb0 = (unsigned)(hi * 32);
            loadt(0, kbb0, bb[0]); loadt(1, kbb0, bb[1]); loadt(2, kbb0, bb[2]);
        }

#pragma unroll 2
        for (int k0 = 0; k0 < HH; k0 += 32) {
            const int kA0 = k0 + (hi ? 16 : 0);
            const int kA1 = k0 + (hi ? 24 : 8);
            BFPack ah, ac, ax;
            ah.q[0] = *(const uint4*)(hrow + kA0);
            ah.q[1] = *(const uint4*)(hrow + kA1);
            ac.q[0] = *(const uint4*)(crow + kA0);
            ac.q[1] = *(const uint4*)(crow + kA1);
            ax.q[0] = *(const uint4*)(xrow + kA0);
            ax.q[1] = *(const uint4*)(xrow + kA1);

            const unsigned kBB  = (unsigned)((k0 + hi * 16) * 2);   // bytes
            const unsigned kBB2 = kBB + 64;   // next k-step (last-step overshoot
                                              // reads stay inside the workspace)
#pragma unroll
            for (int i = 0; i < 18; ++i) {
                if (i < 15) loadt(i + 3, kBB,  bb[(i + 3) % 6]);
                else        loadt(i - 15, kBB2, bb[(i - 15) % 6]);
                const int ai = (i < 10) ? i : (i - 10);
                const v16bf a = (i < 8) ? ah.v : ((i < 10) ? ac.v : ax.v);
                acc[ai] = WMMA_BF16(a, bb[i % 6].v, acc[ai]);
            }
        }

        __syncthreads();   // all waves done reading h/c LDS before overwrite

        // ---- gate math + state update --------------------------------------
        float ttv[8];
#pragma unroll
        for (int r = 0; r < 8; ++r) {
            const int m = r + 8 * hi;
            unsigned tso = (unsigned)(((bm0 + m) * Ss + t) * 4);      // bytes
            ttv[r] = *(const float*)((const char*)tstamps + tso);
        }

#pragma unroll
        for (int j = 0; j < 2; ++j) {
            const int n = wave * 32 + 16 * j + lh;
#pragma unroll
            for (int r = 0; r < 8; ++r) {
                const int m   = r + 8 * hi;
                float cs1  = ftanh(acc[8 + j][r] + bD[j]);
                float cold = creg[j][r];
                float cadj = cold - cs1 + cs1 * ttv[r];
                float fg  = fsigm(acc[0 + 4 * j][r] + bF[j]);
                float ig  = fsigm(acc[1 + 4 * j][r] + bI[j]);
                float og  = fsigm(acc[2 + 4 * j][r] + bO[j]);
                float ctn = ftanh(acc[3 + 4 * j][r] + bC[j]);
                float cn  = fg * cadj + ig * ctn;
                float hn  = og * ftanh(cn);
                creg[j][r] = cn;
                hbf[m * HPITCH + n] = f2bf(hn);
                cbf[m * HPITCH + n] = f2bf(cn);
                // 32-bit byte offset into the 268MB output (fits unsigned);
                // S*H*4 = 2^20 bytes per batch row
                unsigned oboff = ((unsigned)(bm0 + m) << 20) +
                                 (unsigned)((t * HH + n) * 4);
                *(float*)((char*)out + oboff) = og;
                if (t == Ss - 1) {
                    hout[(size_t)(bm0 + m) * HH + n]  = hn;
                    cout_[(size_t)(bm0 + m) * HH + n] = cn;
                }
            }
        }

        asm volatile("s_wait_asynccnt 0" ::: "memory");  // next x buffer landed
        __syncthreads();                                  // h/c visible
    }
}

// ---------------------------------------------------------------------------
// Host launcher
// ---------------------------------------------------------------------------
extern "C" void kernel_launch(void* const* d_in, const int* in_sizes, int n_in,
                              void* d_out, int out_size, void* d_ws, size_t ws_size,
                              hipStream_t stream) {
    (void)in_sizes; (void)n_in; (void)out_size; (void)ws_size;

    const float* inputs   = (const float*)d_in[0];  // [B,S,IN]
    const float* tstamps  = (const float*)d_in[1];  // [B,S]
    const float* Wall_w   = (const float*)d_in[2];  // [4H,H]
    const float* Wall_b   = (const float*)d_in[3];  // [4H]
    const float* U_w      = (const float*)d_in[4];  // [4H,IN]
    const float* U_b      = (const float*)d_in[5];  // [4H]
    const float* Wd_w     = (const float*)d_in[6];  // [H,H]
    const float* Wd_b     = (const float*)d_in[7];  // [H]

    unsigned char* ws = (unsigned char*)d_ws;
    unsigned short* wallbf = (unsigned short*)(ws + 0);        // 2 MB
    unsigned short* ubf    = (unsigned short*)(ws + 2097152);  // 2 MB
    unsigned short* wdbf   = (unsigned short*)(ws + 4194304);  // 512 KB
    float*          biasc  = (float*)(ws + 4718592);           // 8 KB

    float* out   = (float*)d_out;                       // [B,S,H]
    float* hout  = out + (size_t)Bb * Ss * HH;          // [B,H]
    float* cout_ = hout + (size_t)Bb * HH;              // [B,H]

    prep_kernel<<<(G4 * HH + 255) / 256, 256, 0, stream>>>(
        Wall_w, U_w, Wd_w, Wall_b, U_b, wallbf, ubf, wdbf, biasc);

    scan_kernel<<<Bb / 16, 512, SMEM_BYTES, stream>>>(
        inputs, tstamps, wallbf, ubf, wdbf, biasc, Wd_b, out, hout, cout_);
}
